// CLIPRoIHeads_37039797960935
// MI455X (gfx1250) — matrix operand order repaired and checked by reading, not compiled, not run on hardware
//
#include <hip/hip_runtime.h>
#include <hip/hip_bf16.h>

// ---------------------------------------------------------------------------
// CLIPRoIHeads for MI455X (gfx1250, wave32, WMMA + TDM)
//   Stage 0: pack W and features into hi/lo bf16 WMMA-fragment layout (once)
//   Stage 1: logits = feats @ W + bias, split-bf16 (bf16x3) WMMA, f32 acc,
//            TDM (tensor_load_to_lds) double-buffered LDS pipeline
//   Stage 2: softmax + score/size filter -> atomic compaction (<=2048/img)
//   Stage 3: LDS bitonic sort desc -> class-aware NMS -> top-100 emit
// ---------------------------------------------------------------------------

#define B_IMG        8
#define N_PROP       1000
#define D_DIM        1024
#define C_DIM        1201
#define LDC          1216          // padded logits leading dim
#define NTILES       76            // ceil(1201/16)
#define MTILES       63            // ceil(1000/16)
#define KSTEPS       32            // 1024 / 32
#define IMG_H_F      800.0f
#define IMG_W_F      1333.0f
#define SCORE_THRESH 0.05f
#define NMS_THRESH   0.5f
#define DET_PER_IMG  100
#define TOPK         2048

typedef __attribute__((ext_vector_type(16))) __bf16        v16bf;
typedef __attribute__((ext_vector_type(8)))  float         v8f;
typedef __attribute__((ext_vector_type(4)))  unsigned int  v4u;
typedef __attribute__((ext_vector_type(8)))  int           v8i;
typedef __attribute__((ext_vector_type(4)))  int           v4i;

// Fragment-pack layout: [tile][kstep][prec(hi=0,lo=1)][lane][16] bf16
//   one (kstep,prec) slab = 32 lanes * 16 bf16 = 512 bf16 = 1 KB
//   one kstep (hi+lo)     = 1024 bf16 = 2 KB (contiguous -> TDM-friendly)
#define SLAB 512                   // bf16 per [lane][16] slab
#define KSTEP_BF16 1024            // bf16 per kstep (hi+lo)

// ---------------------------------------------------------------------------
// TDM: 1-D tile load Global -> LDS. D# per CDNA5 ISA.
// data_size = 8 B, tile_dim0 = bytes/8, tensor_dim1 = 1, groups 2/3 unused.
// ---------------------------------------------------------------------------
__device__ __forceinline__ void tdm_tile_load(const void* g, void* l, unsigned bytes)
{
#if __has_builtin(__builtin_amdgcn_tensor_load_to_lds)
  const unsigned long long ga  = (unsigned long long)(uintptr_t)g;
  const unsigned           lds = (unsigned)(uintptr_t)l;   // addr[31:0] = LDS offset
  const unsigned           n8  = bytes >> 3;               // 8-byte elements
  v4u g0;
  g0[0] = 1u;                                              // count=1, user mode
  g0[1] = lds;                                             // lds_addr
  g0[2] = (unsigned)ga;                                    // global_addr[31:0]
  g0[3] = ((unsigned)(ga >> 32) & 0x01FFFFFFu) | (2u << 30); // addr[56:32] | type=2
  v8i g1;
  g1[0] = (int)(3u << 16);                                 // data_size = 8 B
  g1[1] = (int)((n8 & 0xFFFFu) << 16);                     // tensor_dim0[15:0]
  g1[2] = (int)(((n8 >> 16) & 0xFFFFu) | (1u << 16));      // tensor_dim0[31:16], dim1=1
  g1[3] = (int)((n8 & 0xFFFFu) << 16);                     // tile_dim0
  g1[4] = 0;                                               // tile_dim1/2 unused
  g1[5] = (int)n8;                                         // tensor_dim0_stride
  g1[6] = 0;
  g1[7] = 0;
  v4i z4 = {};
#if __clang_major__ >= 23
  v8i z8 = {};
  __builtin_amdgcn_tensor_load_to_lds(g0, g1, z4, z4, z8, 0);
#else
  __builtin_amdgcn_tensor_load_to_lds(g0, g1, z4, z4, 0);
#endif
#else
  // fallback: cooperative wave copy (keeps file compiling on any toolchain)
  const int lane = (int)(threadIdx.x & 31);
  const float4* src = (const float4*)g;
  float4* dst = (float4*)l;
  for (unsigned i = lane; i < (bytes >> 4); i += 32) dst[i] = src[i];
#endif
}

// ---------------------------------------------------------------------------
// Stage 0a: pack W[D,C] -> Bpack (hi/lo bf16 fragment order). Done once,
// 5 MB, L2-resident afterwards. B frag map: n = r15, K = half*16 + idx.
// ---------------------------------------------------------------------------
__global__ __launch_bounds__(256) void pack_W(const float* __restrict__ W,
                                              __bf16* __restrict__ Bpack)
{
  const int t = blockIdx.x * 256 + threadIdx.x;
  if (t >= NTILES * KSTEPS * 32 * 16) return;
  const int idx   = t & 15;
  const int lane  = (t >> 4) & 31;
  const int kstep = (t >> 9) & 31;
  const int ntile = t >> 14;
  const int half  = lane >> 4;
  const int n = min(ntile * 16 + (lane & 15), C_DIM - 1);
  const int k = kstep * 32 + half * 16 + idx;
  const float w = W[(size_t)k * C_DIM + n];
  const __bf16 hi = (__bf16)w;
  const __bf16 lo = (__bf16)(w - (float)hi);
  const size_t base = ((size_t)(ntile * KSTEPS + kstep) * 2) * SLAB + lane * 16 + idx;
  Bpack[base]        = hi;
  Bpack[base + SLAB] = lo;
}

// ---------------------------------------------------------------------------
// Stage 0b: pack feats -> Apack. A frag map:
//   K = half*16 + idx,  m = r15
// ---------------------------------------------------------------------------
__global__ __launch_bounds__(256) void pack_A(const float* __restrict__ feats,
                                              __bf16* __restrict__ Apack)
{
  const int t = blockIdx.x * 256 + threadIdx.x;
  if (t >= B_IMG * MTILES * KSTEPS * 32 * 16) return;
  const int idx   = t & 15;
  const int lane  = (t >> 4) & 31;
  const int kstep = (t >> 9) & 31;
  const int rest  = t >> 14;
  const int mtile = rest % MTILES;
  const int img   = rest / MTILES;
  const int half  = lane >> 4;
  const int m = min(mtile * 16 + (lane & 15), N_PROP - 1);
  const int k = kstep * 32 + half * 16 + idx;
  const float a = feats[((size_t)img * N_PROP + m) * D_DIM + k];
  const __bf16 hi = (__bf16)a;
  const __bf16 lo = (__bf16)(a - (float)hi);
  const size_t base = ((size_t)((img * MTILES + mtile) * KSTEPS + kstep) * 2) * SLAB
                    + lane * 16 + idx;
  Apack[base]        = hi;
  Apack[base + SLAB] = lo;
}

// ---------------------------------------------------------------------------
// Stage 1: GEMM. 4 waves/block share an M-tile; wave = one 16x16 N-tile.
// Double-buffered LDS pipeline: per chunk (2 K-steps = 4 KB/stream), wave 0
// TDM-loads the shared A chunk, every wave TDM-loads its own B chunk, then
// 2 x (3 WMMA) consume ds_load_b128 fragments. Sync = s_wait_tensorcnt + bar.
// ---------------------------------------------------------------------------
__global__ __launch_bounds__(128) void gemm_logits_wmma(
    const __bf16* __restrict__ Apack,
    const __bf16* __restrict__ Bpack,
    const float*  __restrict__ bias,
    float* __restrict__ logits)        // [B, N_PROP, LDC]
{
  __shared__ __align__(32) __bf16 sA[2][2][2][32][16];      //  8 KB [buf][ks][prec]
  __shared__ __align__(32) __bf16 sB[2][4][2][2][32][16];   // 32 KB [buf][wave][ks][prec]

  const int lane  = threadIdx.x & 31;
  const int wave  = threadIdx.x >> 5;
  const int ntile = blockIdx.x * 4 + wave;   // 0..75
  const int mtile = blockIdx.y;              // 0..62
  const int img   = blockIdx.z;
  const int r15   = lane & 15;
  const int half  = lane >> 4;

  const __bf16* aBase = Apack + (size_t)((img * MTILES + mtile) * KSTEPS) * KSTEP_BF16;
  const __bf16* bBase = Bpack + (size_t)(ntile * KSTEPS) * KSTEP_BF16;

  // prologue: chunk 0 -> buffer 0
  if (wave == 0) tdm_tile_load(aBase, &sA[0][0][0][0][0], 4096);
  tdm_tile_load(bBase, &sB[0][wave][0][0][0][0], 4096);
  __builtin_amdgcn_s_wait_tensorcnt(0);
  __syncthreads();

  v8f acc = {};
  for (int c = 0; c < KSTEPS / 2; ++c) {
    const int cur = c & 1, nxt = cur ^ 1;
    if (c + 1 < KSTEPS / 2) {                       // prefetch next chunk via TDM
      const size_t off = (size_t)(c + 1) * 2 * KSTEP_BF16;
      if (wave == 0) tdm_tile_load(aBase + off, &sA[nxt][0][0][0][0], 4096);
      tdm_tile_load(bBase + off, &sB[nxt][wave][0][0][0][0], 4096);
    }
#pragma unroll
    for (int ks = 0; ks < 2; ++ks) {
      const v16bf ahi = *(const v16bf*)&sA[cur][ks][0][lane][0];
      const v16bf alo = *(const v16bf*)&sA[cur][ks][1][lane][0];
      const v16bf bhi = *(const v16bf*)&sB[cur][wave][ks][0][lane][0];
      const v16bf blo = *(const v16bf*)&sB[cur][wave][ks][1][lane][0];
      acc = __builtin_amdgcn_wmma_f32_16x16x32_bf16(false, ahi, false, bhi, (short)0, acc, false, false);
      acc = __builtin_amdgcn_wmma_f32_16x16x32_bf16(false, alo, false, bhi, (short)0, acc, false, false);
      acc = __builtin_amdgcn_wmma_f32_16x16x32_bf16(false, ahi, false, blo, (short)0, acc, false, false);
    }
    __builtin_amdgcn_s_wait_tensorcnt(0);
    __syncthreads();
  }

  // C/D layout: acc[r] = C[M = half*8 + r][N = r15]
  const int gn = blockIdx.x * 64 + wave * 16 + r15;
  if (gn < C_DIM) {
    const float bv = bias[gn];
#pragma unroll
    for (int r = 0; r < 8; ++r) {
      const int gm = mtile * 16 + (half << 3) + r;
      if (gm < N_PROP)
        logits[((size_t)img * N_PROP + gm) * LDC + gn] = acc[r] + bv;
    }
  }
}

// ---------------------------------------------------------------------------
// zero the candidate buffer + per-image counters (graph-safe, no memset API)
// ---------------------------------------------------------------------------
__global__ void init_ws(unsigned long long* __restrict__ cand,
                        int* __restrict__ counters)
{
  const int t = blockIdx.x * blockDim.x + threadIdx.x;
  if (t < B_IMG * TOPK) cand[t] = 0ull;
  if (t < B_IMG)        counters[t] = 0;
}

// ---------------------------------------------------------------------------
// Stage 2: one wave per proposal. Softmax over C=1201, filter score>0.05 and
// clipped-size>=0.01, compact (score, prop*2048+cls) into per-image buffer.
// ---------------------------------------------------------------------------
__global__ __launch_bounds__(256) void softmax_compact(
    const float* __restrict__ logits,       // [B, N_PROP, LDC]
    const float* __restrict__ proposals,    // [B, N_PROP, 4]
    unsigned long long* __restrict__ cand,  // [B, TOPK]
    int* __restrict__ counters)             // [B]
{
  const int lane = threadIdx.x & 31;
  const int wave = threadIdx.x >> 5;
  const int p    = blockIdx.x * 8 + wave;
  const int img  = blockIdx.y;
  if (p >= N_PROP) return;

  const float* __restrict__ row = logits + ((size_t)img * N_PROP + p) * LDC;

  float m = -3.4e38f;
  for (int c = lane; c < C_DIM; c += 32) m = fmaxf(m, row[c]);
#pragma unroll
  for (int s = 16; s > 0; s >>= 1) m = fmaxf(m, __shfl_xor(m, s, 32));

  float sum = 0.0f;
  for (int c = lane; c < C_DIM; c += 32) sum += __expf(row[c] - m);
#pragma unroll
  for (int s = 16; s > 0; s >>= 1) sum += __shfl_xor(sum, s, 32);
  const float inv = 1.0f / sum;

  const float4 pr = *(const float4*)(proposals + ((size_t)img * N_PROP + p) * 4);
  const float x1 = fminf(fmaxf(pr.x, 0.0f), IMG_W_F);
  const float y1 = fminf(fmaxf(pr.y, 0.0f), IMG_H_F);
  const float x2 = fminf(fmaxf(pr.z, 0.0f), IMG_W_F);
  const float y2 = fminf(fmaxf(pr.w, 0.0f), IMG_H_F);
  if (((x2 - x1) < 0.01f) || ((y2 - y1) < 0.01f)) return;   // remove_small_boxes

  for (int c = 1 + lane; c < C_DIM; c += 32) {              // skip background
    const float score = __expf(row[c] - m) * inv;
    if (score > SCORE_THRESH) {
      const int pos = atomicAdd(&counters[img], 1);
      if (pos < TOPK) {
        const unsigned int fid = ((unsigned)p << 11) | (unsigned)c;
        cand[(size_t)img * TOPK + pos] =
            ((unsigned long long)__float_as_uint(score) << 32) | (unsigned long long)fid;
      }
    }
  }
}

// ---------------------------------------------------------------------------
// Stage 3: one block/image. LDS bitonic sort (desc) of 2048 keys, decode +
// clip boxes into LDS, greedy class-aware NMS with LDS keep bitmask, emit
// first 100 survivors. d_out = [boxes 8*100*4 | scores 8*100 | labels 8*100].
// ---------------------------------------------------------------------------
__global__ __launch_bounds__(256) void nms_topk_out(
    const unsigned long long* __restrict__ cand,
    const float* __restrict__ proposals,
    float* __restrict__ out)
{
  __shared__ unsigned long long key[TOPK];   // 16 KB
  __shared__ float4             bx[TOPK];    // 32 KB
  __shared__ unsigned int       keepm[TOPK / 32];
  __shared__ int                s_cnt;

  const int tid = threadIdx.x;
  const int img = blockIdx.x;

  for (int t = tid; t < TOPK; t += 256) key[t] = cand[(size_t)img * TOPK + t];
  if (tid == 0) s_cnt = 0;
  __syncthreads();

  // bitonic sort, descending (scores > 0 so u64 key order == score order)
  for (int k = 2; k <= TOPK; k <<= 1) {
    for (int j = k >> 1; j > 0; j >>= 1) {
      for (int t = tid; t < TOPK; t += 256) {
        const int ixj = t ^ j;
        if (ixj > t) {
          const unsigned long long a = key[t], b = key[ixj];
          const bool desc = ((t & k) == 0);
          if (desc ? (a < b) : (a > b)) { key[t] = b; key[ixj] = a; }
        }
      }
      __syncthreads();
    }
  }

  // decode + clip boxes; count valid (sorted => valid are the first cnt)
  for (int t = tid; t < TOPK; t += 256) {
    const unsigned long long kv = key[t];
    const unsigned fid = (unsigned)kv;
    const int p = (int)(fid >> 11);
    const float4 pr = *(const float4*)(proposals + ((size_t)img * N_PROP + p) * 4);
    float4 b;
    b.x = fminf(fmaxf(pr.x, 0.0f), IMG_W_F);
    b.y = fminf(fmaxf(pr.y, 0.0f), IMG_H_F);
    b.z = fminf(fmaxf(pr.z, 0.0f), IMG_W_F);
    b.w = fminf(fmaxf(pr.w, 0.0f), IMG_H_F);
    bx[t] = b;
    if (__uint_as_float((unsigned)(kv >> 32)) > SCORE_THRESH) atomicAdd(&s_cnt, 1);
  }
  for (int t = tid; t < TOPK / 32; t += 256) keepm[t] = 0xFFFFFFFFu;
  __syncthreads();
  const int cnt = s_cnt;

  // greedy NMS: same class + IoU > 0.5 suppresses (== offset-coordinate trick)
  for (int i = 0; i < cnt; ++i) {
    if ((keepm[i >> 5] >> (i & 31)) & 1u) {
      const float4 bi = bx[i];
      const unsigned ci = (unsigned)key[i] & 2047u;
      const float ai = (bi.z - bi.x) * (bi.w - bi.y);
      for (int j = i + 1 + tid; j < cnt; j += 256) {
        if (((keepm[j >> 5] >> (j & 31)) & 1u) &&
            (((unsigned)key[j] & 2047u) == ci)) {
          const float4 bj = bx[j];
          const float iw = fmaxf(fminf(bi.z, bj.z) - fmaxf(bi.x, bj.x), 0.0f);
          const float ih = fmaxf(fminf(bi.w, bj.w) - fmaxf(bi.y, bj.y), 0.0f);
          const float inter = iw * ih;
          const float uni = ai + (bj.z - bj.x) * (bj.w - bj.y) - inter;
          if (uni > 0.0f && inter > NMS_THRESH * uni)
            atomicAnd(&keepm[j >> 5], ~(1u << (j & 31)));
        }
      }
    }
    __syncthreads();
  }

  float* det_boxes  = out;                                   // [B,100,4]
  float* det_scores = out + (size_t)B_IMG * DET_PER_IMG * 4; // [B,100]
  int*   det_labels = (int*)(out + (size_t)B_IMG * DET_PER_IMG * 5); // [B,100]

  for (int t = tid; t < DET_PER_IMG * 4; t += 256) det_boxes[img * DET_PER_IMG * 4 + t] = 0.0f;
  for (int t = tid; t < DET_PER_IMG; t += 256) {
    det_scores[img * DET_PER_IMG + t] = 0.0f;
    det_labels[img * DET_PER_IMG + t] = -1;
  }
  __syncthreads();

  if (tid == 0) {
    int nout = 0;
    for (int t = 0; t < cnt && nout < DET_PER_IMG; ++t) {
      if ((keepm[t >> 5] >> (t & 31)) & 1u) {
        const unsigned long long kv = key[t];
        const float4 b = bx[t];
        float* db = det_boxes + (img * DET_PER_IMG + nout) * 4;
        db[0] = b.x; db[1] = b.y; db[2] = b.z; db[3] = b.w;
        det_scores[img * DET_PER_IMG + nout] = __uint_as_float((unsigned)(kv >> 32));
        det_labels[img * DET_PER_IMG + nout] = (int)((unsigned)kv & 2047u);
        ++nout;
      }
    }
  }
}

// ---------------------------------------------------------------------------
extern "C" void kernel_launch(void* const* d_in, const int* in_sizes, int n_in,
                              void* d_out, int out_size, void* d_ws, size_t ws_size,
                              hipStream_t stream) {
  const float* feats = (const float*)d_in[0];   // [8,1000,1024]
  const float* W     = (const float*)d_in[1];   // [1024,1201]
  const float* bias  = (const float*)d_in[2];   // [1201]
  const float* props = (const float*)d_in[3];   // [8,1000,4]
  float* out = (float*)d_out;

  char* ws = (char*)d_ws;
  const size_t logits_bytes = (size_t)B_IMG * N_PROP * LDC * sizeof(float);            // 38.9 MB
  const size_t apack_bytes  = (size_t)B_IMG * MTILES * KSTEPS * KSTEP_BF16 * 2;        // 33.0 MB
  const size_t bpack_bytes  = (size_t)NTILES * KSTEPS * KSTEP_BF16 * 2;                //  5.0 MB

  float*              logits   = (float*)ws;
  __bf16*             Apack    = (__bf16*)(ws + logits_bytes);
  __bf16*             Bpack    = (__bf16*)(ws + logits_bytes + apack_bytes);
  unsigned long long* cand     = (unsigned long long*)(ws + logits_bytes + apack_bytes + bpack_bytes);
  int*                counters = (int*)(ws + logits_bytes + apack_bytes + bpack_bytes
                                        + (size_t)B_IMG * TOPK * 8);

  init_ws<<<(B_IMG * TOPK + 255) / 256, 256, 0, stream>>>(cand, counters);
  pack_W<<<(NTILES * KSTEPS * 32 * 16) / 256, 256, 0, stream>>>(W, Bpack);
  pack_A<<<(B_IMG * MTILES * KSTEPS * 32 * 16) / 256, 256, 0, stream>>>(feats, Apack);
  gemm_logits_wmma<<<dim3(19, MTILES, B_IMG), 128, 0, stream>>>(Apack, Bpack, bias, logits);
  softmax_compact<<<dim3((N_PROP + 7) / 8, B_IMG), 256, 0, stream>>>(logits, props, cand, counters);
  nms_topk_out<<<B_IMG, 256, 0, stream>>>(cand, props, out);
}
